// _netQ_3238405341233
// MI455X (gfx1250) — matrix operand order repaired and checked by reading, not compiled, not run on hardware
//
#include <hip/hip_runtime.h>
#include <cmath>
#include <cstdint>

#ifndef M_PI
#define M_PI 3.14159265358979323846
#endif

typedef __bf16 bf16;
typedef __attribute__((ext_vector_type(16))) __bf16 v16bf;
typedef __attribute__((ext_vector_type(8)))  __bf16 v8bf;
typedef __attribute__((ext_vector_type(8)))  float  v8f;
typedef __attribute__((ext_vector_type(4))) unsigned int v4u;
typedef __attribute__((ext_vector_type(8))) int v8i;
typedef __attribute__((ext_vector_type(4))) int v4i;

// TDM availability (device pass only; host pass sees 0 and the fallback branch)
#if defined(__has_builtin)
#if __has_builtin(__builtin_amdgcn_tensor_load_to_lds) && __has_builtin(__builtin_amdgcn_s_wait_tensorcnt)
#define HAVE_TDM 1
#endif
#endif
#ifndef HAVE_TDM
#define HAVE_TDM 0
#endif

// ---------------- bf16 convert (round-to-nearest-even) ----------------
__device__ __forceinline__ bf16 f2bf(float f) {
  union { float f; uint32_t u; } c; c.f = f;
  uint32_t u = c.u + 0x7FFFu + ((c.u >> 16) & 1u);
  uint16_t h = (uint16_t)(u >> 16);
  bf16 r; __builtin_memcpy(&r, &h, 2);
  return r;
}

// ---------------- deterministic hash RNG -> normal ----------------
__device__ __forceinline__ unsigned hash2(unsigned a, unsigned b) {
  unsigned x = a * 0x9E3779B1u ^ (b + 0x85EBCA77u) * 0xC2B2AE3Du;
  x ^= x >> 16; x *= 0x7FEB352Du; x ^= x >> 15; x *= 0x846CA68Bu; x ^= x >> 16;
  return x;
}
__device__ __forceinline__ float u01(unsigned h) {
  return ((float)(h >> 8) + 0.5f) * (1.0f / 16777216.0f);
}
__device__ __forceinline__ float rnorm(unsigned seed, unsigned idx) {
  float u1 = u01(hash2(seed, 2u * idx + 1u));
  float u2 = u01(hash2(seed ^ 0xB5297A4Du, 2u * idx + 2u));
  return sqrtf(-2.0f * logf(u1)) * cosf(6.28318530718f * u2);
}

// ---------------- NCHW f32 -> NHWC bf16 ----------------
__global__ void k_convert_x(const float* __restrict__ x, bf16* __restrict__ out, int total) {
  int i = blockIdx.x * 256 + threadIdx.x;
  if (i >= total) return;
  int c = i % 3; int p = i / 3;
  int xw = p & 31; p >>= 5;
  int y = p & 31; int n = p >> 5;
  out[i] = f2bf(x[(((size_t)n * 3 + c) * 32 + y) * 32 + xw]);
}

// ---------------- weight swizzle: linear (rows = out neuron) ----------------
// panel element layout (per 16x16x32 fragment): lane l, half h:
//   N = nt*16 + (l&15), K = kc*32 + 8*(l>>4) + (h<8 ? h : h+8)
__global__ void k_swz_lin(const float* __restrict__ Wsrc, int rowlen, int col0, int Kdim,
                          int KC, bf16* __restrict__ pan, long total) {
  long id = (long)blockIdx.x * 256 + threadIdx.x;
  if (id >= total) return;
  int e = (int)(id & 511); long t = id >> 9;
  int kc = (int)(t % KC); int nt = (int)(t / KC);
  int l = e >> 4, h = e & 15;
  int n = nt * 16 + (l & 15);
  int k = kc * 32 + 8 * (l >> 4) + (h < 8 ? h : h + 8);
  float v = (k < Kdim) ? Wsrc[(size_t)n * rowlen + col0 + k] : 0.0f;
  pan[id] = f2bf(v);
}

// ---------------- weight swizzle: conv (OIHW -> im2col K=(dy,dx,c)) ----------------
__global__ void k_swz_conv(const float* __restrict__ Wsrc, int CI, int KH, int KW, int Ktotal,
                           int KC, bf16* __restrict__ pan, long total) {
  long id = (long)blockIdx.x * 256 + threadIdx.x;
  if (id >= total) return;
  int e = (int)(id & 511); long t = id >> 9;
  int kc = (int)(t % KC); int nt = (int)(t / KC);
  int l = e >> 4, h = e & 15;
  int n = nt * 16 + (l & 15);
  int k = kc * 32 + 8 * (l >> 4) + (h < 8 ? h : h + 8);
  float v = 0.0f;
  if (k < Ktotal) {
    int c = k % CI; int t2 = k / CI;
    int dx = t2 % KW; int dy = t2 / KW;
    v = Wsrc[(((size_t)n * CI + c) * KH + dy) * KW + dx];
  }
  pan[id] = f2bf(v);
}

// ---------------- implicit-GEMM conv, bf16 WMMA, fused bias+leaky ----------------
// All geometry compile-time: divisions become shifts/constant muls.
template <int CI, int CO, int KK, int S, int P, int IH, int OH, int KC, int KTOT, int LEAKY>
__global__ void k_conv_gemm(const bf16* __restrict__ A, const bf16* __restrict__ pan,
                            const float* __restrict__ bias, bf16* __restrict__ out) {
  const int lane = threadIdx.x & 31;
  const int wv   = threadIdx.x >> 5;
  const int mt = blockIdx.x * 4 + wv;
  const int nt = blockIdx.y;
  const int hi = lane >> 4;
  const int mrow = mt * 16 + (lane & 15);
  constexpr int OHW = OH * OH;
  const int nimg = mrow / OHW;
  const int rem = mrow - nimg * OHW;
  const int oy = rem / OH;
  const int ox = rem - oy * OH;
  const bf16* panw = pan + (size_t)nt * KC * 512 + lane * 16;
  const bf16 zb = f2bf(0.0f);
  v8f acc = {};
  for (int kc = 0; kc < KC; ++kc) {
    v16bf b = *(const v16bf*)(panw + (size_t)kc * 512);
    v16bf a;
    if constexpr ((CI % 32) == 0) {
      const int k0 = kc * 32;
      const int c0 = k0 % CI;
      const int cell = k0 / CI;
      const int dx = cell % KK, dy = cell / KK;
      const int iy = oy * S - P + dy;
      const int ix = ox * S - P + dx;
      if ((unsigned)iy < (unsigned)IH && (unsigned)ix < (unsigned)IH) {
        const bf16* row = A + ((size_t)(nimg * IH + iy) * IH + ix) * CI + c0 + 8 * hi;
        v8bf lo = *(const v8bf*)(row);
        v8bf h8 = *(const v8bf*)(row + 16);
#pragma unroll
        for (int q = 0; q < 8; ++q) { a[q] = lo[q]; a[q + 8] = h8[q]; }
      } else {
#pragma unroll
        for (int q = 0; q < 16; ++q) a[q] = zb;
      }
    } else {
#pragma unroll
      for (int h = 0; h < 16; ++h) {
        const int koff = 8 * hi + (h < 8 ? h : h + 8);
        const int k = kc * 32 + koff;
        bf16 v = zb;
        if (k < KTOT) {
          const int c = k % CI; const int t2 = k / CI;
          const int dx = t2 % KK; const int dy = t2 / KK;
          const int iy = oy * S - P + dy;
          const int ix = ox * S - P + dx;
          if ((unsigned)iy < (unsigned)IH && (unsigned)ix < (unsigned)IH)
            v = A[((size_t)(nimg * IH + iy) * IH + ix) * CI + c];
        }
        a[h] = v;
      }
    }
    if (kc + 1 < KC) __builtin_prefetch(panw + (size_t)(kc + 1) * 512, 0, 0);
    acc = __builtin_amdgcn_wmma_f32_16x16x32_bf16(false, a, false, b, (short)0, acc, false, false);
  }
  const int n = nt * 16 + (lane & 15);
  const float bv = bias[n];
#pragma unroll
  for (int v = 0; v < 8; ++v) {
    int m = mt * 16 + v + 8 * hi;
    float o = acc[v] + bv;
    if (LEAKY) o = (o >= 0.0f) ? o : 0.2f * o;
    out[(size_t)m * CO + n] = f2bf(o);
  }
}

// ---------------- dense GEMM, bf16 WMMA, TDM-staged B panel, fused CSL epilogue ----------------
// mode 0: plain f32 store (xemb @ Wg_x.T precompute)
// mode 1: ConcatSquashLinear: o = (acc+b)*sigmoid(xg+gv) + xb + bv, optional leaky 0.01
// B panel for this block's n-tile (KC*1024 bytes, <=16KB) is DMA'd into LDS by the
// Tensor Data Mover (wave 0 issues descriptor, waits tensorcnt, then block barrier).
__global__ void k_gemm_lin(const bf16* __restrict__ A, int lda,
                           const bf16* __restrict__ pan, int KC,
                           float* __restrict__ outf, bf16* __restrict__ outb, int ldc,
                           const float* __restrict__ xg, const float* __restrict__ xb,
                           const float* __restrict__ gv, const float* __restrict__ bv,
                           const float* __restrict__ bias, int mode, int leaky) {
  const int lane = threadIdx.x & 31;
  const int wv   = threadIdx.x >> 5;
  const int mt = blockIdx.x * 4 + wv;
  const int nt = blockIdx.y;
  const int hi = lane >> 4;
  const int mrow = mt * 16 + (lane & 15);
  const bf16* arow = A + (size_t)mrow * lda + 8 * hi;

#if HAVE_TDM
  __shared__ __align__(32) bf16 sB[16 * 512];  // up to KC=16 panel chunks
  if (wv == 0) {
    const unsigned ldsAddr = (unsigned)(size_t)(void*)sB;
    const unsigned long long ga =
        (unsigned long long)(size_t)(const void*)(pan + (size_t)nt * KC * 512);
    const unsigned elems = (unsigned)KC * 512u;  // bf16 elements, 1-D tile
    v4u g0;
    g0[0] = 1u;                                   // count=1 valid user descriptor
    g0[1] = ldsAddr;                              // lds_addr
    g0[2] = (unsigned)(ga & 0xFFFFFFFFu);         // global_addr[31:0]
    g0[3] = (unsigned)((ga >> 32) & 0x01FFFFFFu) | (2u << 30);  // addr[56:32] | type=2
    v8i g1;
    g1[0] = (int)(1u << 16);                      // workgroup_mask=0, data_size=1 (2B)
    g1[1] = (int)((elems & 0xFFFFu) << 16);       // tensor_dim0[15:0]
    g1[2] = (int)((elems >> 16) | (1u << 16));    // tensor_dim0[31:16] | tensor_dim1=1
    g1[3] = (int)((elems & 0xFFFFu) << 16);       // tile_dim0 = elems
    g1[4] = 1;                                    // tile_dim1=1, tile_dim2=0
    g1[5] = (int)elems;                           // tensor_dim0_stride[31:0]
    g1[6] = 0;
    g1[7] = 0;
    v4i z4 = {0, 0, 0, 0};
#if defined(__clang_major__) && (__clang_major__ >= 23)
    v8i z8 = {0, 0, 0, 0, 0, 0, 0, 0};
    __builtin_amdgcn_tensor_load_to_lds(g0, g1, z4, z4, z8, 0);
#else
    __builtin_amdgcn_tensor_load_to_lds(g0, g1, z4, z4, 0);
#endif
    __builtin_amdgcn_s_wait_tensorcnt(0);
  }
  __syncthreads();
#else
  const bf16* panw = pan + (size_t)nt * KC * 512 + lane * 16;
#endif

  v8f acc = {};
  for (int kc = 0; kc < KC; ++kc) {
#if HAVE_TDM
    v16bf b = *(const v16bf*)(sB + (size_t)kc * 512 + lane * 16);
#else
    v16bf b = *(const v16bf*)(panw + (size_t)kc * 512);
#endif
    v8bf lo = *(const v8bf*)(arow + kc * 32);
    v8bf h8 = *(const v8bf*)(arow + kc * 32 + 16);
    v16bf a;
#pragma unroll
    for (int q = 0; q < 8; ++q) { a[q] = lo[q]; a[q + 8] = h8[q]; }
    if (kc + 1 < KC) __builtin_prefetch(arow + (kc + 1) * 32, 0, 0);
    acc = __builtin_amdgcn_wmma_f32_16x16x32_bf16(false, a, false, b, (short)0, acc, false, false);
  }
  const int n = nt * 16 + (lane & 15);
  if (mode == 0) {
#pragma unroll
    for (int v = 0; v < 8; ++v) {
      int m = mt * 16 + v + 8 * hi;
      outf[(size_t)m * ldc + n] = acc[v];
    }
  } else {
    const float bb = bias[n], g0v = gv[n], b0v = bv[n];
#pragma unroll
    for (int v = 0; v < 8; ++v) {
      int m = mt * 16 + v + 8 * hi;
      float pre = acc[v] + bb;
      float gate = 1.0f / (1.0f + __expf(-(xg[(size_t)m * ldc + n] + g0v)));
      float o = pre * gate + xb[(size_t)m * ldc + n] + b0v;
      if (leaky) o = (o >= 0.0f) ? o : 0.01f * o;
      outb[(size_t)m * ldc + n] = f2bf(o);
      if (outf) outf[(size_t)m * ldc + n] = o;
    }
  }
}

// ---------------- per-step temb MLP + temb-part of gate/bias vectors ----------------
struct TembArgs {
  const float *W0, *b0, *W1, *b1;
  const float *Wg[6], *bg[6], *Wb[6];
  float *gvec, *bvec;
  float tt[20];
  int dox[6], doff[6];
};

__global__ void k_temb(TembArgs a) {
  __shared__ float s0[128], s1[128], s2[128];
  const int s = blockIdx.x;
  const int j = threadIdx.x;
  const float tt = a.tt[s];
  int hf = j & 63;
  float f = expf((float)hf * (-logf(10000.0f) / 63.0f));
  float ang = tt * f;
  s0[j] = (j >= 64) ? cosf(ang) : sinf(ang);
  __syncthreads();
  float acc = a.b0[j];
  for (int k = 0; k < 128; ++k) acc += s0[k] * a.W0[(size_t)j * 128 + k];
  s1[j] = acc / (1.0f + __expf(-acc));
  __syncthreads();
  acc = a.b1[j];
  for (int k = 0; k < 128; ++k) acc += s1[k] * a.W1[(size_t)j * 128 + k];
  s2[j] = acc;
  __syncthreads();
  for (int l = 0; l < 6; ++l) {
    int dl = a.dox[l];
    for (int n = j; n < dl; n += 128) {
      float g = a.bg[l][n];
      float bsum = 0.0f;
      const float* wg = a.Wg[l] + (size_t)n * 256;
      const float* wb = a.Wb[l] + (size_t)n * 256;
      for (int k = 0; k < 128; ++k) { g += s2[k] * wg[k]; bsum += s2[k] * wb[k]; }
      a.gvec[(size_t)s * 1408 + a.doff[l] + n] = g;
      a.bvec[(size_t)s * 1408 + a.doff[l] + n] = bsum;
    }
  }
}

// ---------------- zt init / per-step reverse diffusion update ----------------
__global__ void k_init_z(float* __restrict__ zt, bf16* __restrict__ ztb, unsigned seed, int total) {
  int i = blockIdx.x * 256 + threadIdx.x;
  if (i >= total) return;
  float v = rnorm(seed, (unsigned)i);
  zt[i] = v; ztb[i] = f2bf(v);
}

__global__ void k_update_z(const float* __restrict__ eps, float* __restrict__ zt,
                           bf16* __restrict__ ztb, float* __restrict__ out,
                           float c1, float c2, float ra, float ob, float stdv,
                           int last, unsigned seed, int total) {
  int i = blockIdx.x * 256 + threadIdx.x;
  if (i >= total) return;
  float z = zt[i];
  float pz = c1 * (z - eps[i] * c2);
  if (last) { out[i] = pz; return; }
  float mean = ra * z + ob * pz;
  float nz = mean + stdv * rnorm(seed, (unsigned)i);
  zt[i] = nz; ztb[i] = f2bf(nz);
}

// =====================================================================
extern "C" void kernel_launch(void* const* d_in, const int* in_sizes, int n_in,
                              void* d_out, int out_size, void* d_ws, size_t ws_size,
                              hipStream_t stream) {
  static const long encWsz[5] = {1728, 131072, 524288, 2097152, 1048576};
  static const long encBsz[5] = {64, 128, 256, 512, 128};
  static const long cslWsz[6] = {16384, 32768, 131072, 131072, 32768, 16384};
  static const long cslBsz[6] = {128, 256, 512, 256, 128, 128};
  static const long cslGsz[6] = {32768, 65536, 131072, 65536, 32768, 32768};
  const long Xsz = 2048L * 3 * 32 * 32;

  const float *pEncW[5], *pEncB[5], *pTW[2], *pTB[2];
  const float *pW[6], *pB[6], *pWg[6], *pBg[6], *pWb[6], *pX = nullptr;

  struct Slot { const float** p; long sz; };
  Slot ord[48];

  auto build0 = [&](Slot* o) {  // reference dict insertion order
    int c = 0;
    o[c++] = {&pX, Xsz};
    for (int l = 0; l < 5; ++l) { o[c++] = {&pEncW[l], encWsz[l]}; o[c++] = {&pEncB[l], encBsz[l]}; }
    for (int l = 0; l < 2; ++l) { o[c++] = {&pTW[l], 16384}; o[c++] = {&pTB[l], 128}; }
    for (int l = 0; l < 6; ++l) {
      o[c++] = {&pW[l], cslWsz[l]}; o[c++] = {&pB[l], cslBsz[l]};
      o[c++] = {&pWg[l], cslGsz[l]}; o[c++] = {&pBg[l], cslBsz[l]}; o[c++] = {&pWb[l], cslGsz[l]};
    }
    return c;
  };
  auto build1 = [&](Slot* o) {  // x first, JAX-sorted inner dicts
    int c = 0;
    o[c++] = {&pX, Xsz};
    for (int l = 0; l < 6; ++l) {
      o[c++] = {&pW[l], cslWsz[l]}; o[c++] = {&pWb[l], cslGsz[l]}; o[c++] = {&pWg[l], cslGsz[l]};
      o[c++] = {&pB[l], cslBsz[l]}; o[c++] = {&pBg[l], cslBsz[l]};
    }
    for (int l = 0; l < 5; ++l) { o[c++] = {&pEncW[l], encWsz[l]}; o[c++] = {&pEncB[l], encBsz[l]}; }
    for (int l = 0; l < 2; ++l) { o[c++] = {&pTW[l], 16384}; o[c++] = {&pTB[l], 128}; }
    return c;
  };
  auto build2 = [&](Slot* o) {  // fully sorted pytree, x last
    int c = 0;
    for (int l = 0; l < 6; ++l) {
      o[c++] = {&pW[l], cslWsz[l]}; o[c++] = {&pWb[l], cslGsz[l]}; o[c++] = {&pWg[l], cslGsz[l]};
      o[c++] = {&pB[l], cslBsz[l]}; o[c++] = {&pBg[l], cslBsz[l]};
    }
    for (int l = 0; l < 5; ++l) { o[c++] = {&pEncW[l], encWsz[l]}; o[c++] = {&pEncB[l], encBsz[l]}; }
    for (int l = 0; l < 2; ++l) { o[c++] = {&pTW[l], 16384}; o[c++] = {&pTB[l], 128}; }
    o[c++] = {&pX, Xsz};
    return c;
  };

  bool matched = false;
  for (int v = 0; v < 3 && !matched; ++v) {
    int c = (v == 0) ? build0(ord) : (v == 1) ? build1(ord) : build2(ord);
    if (c != n_in) continue;
    bool ok = true;
    for (int i = 0; i < c; ++i)
      if ((long)in_sizes[i] != ord[i].sz) { ok = false; break; }
    if (ok) {
      for (int i = 0; i < c; ++i) *ord[i].p = (const float*)d_in[i];
      matched = true;
    }
  }
  if (!matched) {
    int c = build0(ord);
    if (c > n_in) return;
    for (int i = 0; i < c; ++i) *ord[i].p = (const float*)d_in[i];
  }

  // ---- workspace layout ----
  size_t off = 0;
  auto alloc = [&](size_t b) { size_t o = off; off = (off + b + 255) & ~(size_t)255; return o; };
  uint8_t* W = (uint8_t*)d_ws;

  const int convCI[5] = {3, 64, 128, 256, 512};
  const int convCO[5] = {64, 128, 256, 512, 128};
  const int convK[5]  = {3, 4, 4, 4, 4};
  int convKtot[5], convKC[5];
  size_t o_xnhwc = alloc((size_t)Xsz * 2);
  size_t o_cpan[5];
  for (int l = 0; l < 5; ++l) {
    convKtot[l] = convCI[l] * convK[l] * convK[l];
    convKC[l] = (convKtot[l] + 31) / 32;
    o_cpan[l] = alloc((size_t)convKC[l] * (convCO[l] / 16) * 512 * 2);
  }
  const int di[6] = {128, 128, 256, 512, 256, 128};
  const int dd[6] = {128, 256, 512, 256, 128, 128};
  int doff[6]; { int s = 0; for (int l = 0; l < 6; ++l) { doff[l] = s; s += dd[l]; } }
  size_t o_wpan[6], o_gpan[6], o_bpan[6];
  for (int l = 0; l < 6; ++l) {
    o_wpan[l] = alloc((size_t)(di[l] / 32) * (dd[l] / 16) * 512 * 2);
    o_gpan[l] = alloc((size_t)4 * (dd[l] / 16) * 512 * 2);
    o_bpan[l] = alloc((size_t)4 * (dd[l] / 16) * 512 * 2);
  }
  size_t o_xemb = alloc(2048L * 128 * 2);
  size_t o_xg = alloc(2048L * 1408 * 4);
  size_t o_xb = alloc(2048L * 1408 * 4);
  size_t o_gv = alloc(20L * 1408 * 4);
  size_t o_bv = alloc(20L * 1408 * 4);
  size_t o_zt = alloc(2048L * 128 * 4);
  size_t o_ztb = alloc(2048L * 128 * 2);
  size_t o_eps = alloc(2048L * 128 * 4);
  size_t o_h0 = alloc(2048L * 512 * 2);
  size_t o_h1 = alloc(2048L * 512 * 2);
  size_t o_ra = alloc((size_t)512 * 16 * 16 * 128 * 2);
  size_t o_rb = alloc((size_t)512 * 32 * 32 * 64 * 2);
  if (off > ws_size) return;

  bf16* xnhwc = (bf16*)(W + o_xnhwc);
  bf16* xembbf = (bf16*)(W + o_xemb);
  float* xg = (float*)(W + o_xg);
  float* xb = (float*)(W + o_xb);
  float* gvec = (float*)(W + o_gv);
  float* bvec = (float*)(W + o_bv);
  float* zt = (float*)(W + o_zt);
  bf16* ztb = (bf16*)(W + o_ztb);
  float* eps = (float*)(W + o_eps);
  bf16* h0 = (bf16*)(W + o_h0);
  bf16* h1 = (bf16*)(W + o_h1);
  bf16* regA = (bf16*)(W + o_ra);
  bf16* regB = (bf16*)(W + o_rb);

  // ---- stage 0: input + weight conversion/swizzle ----
  k_convert_x<<<(int)((Xsz + 255) / 256), 256, 0, stream>>>(pX, xnhwc, (int)Xsz);
  for (int l = 0; l < 5; ++l) {
    long tot = (long)convKC[l] * (convCO[l] / 16) * 512;
    k_swz_conv<<<(int)((tot + 255) / 256), 256, 0, stream>>>(
        pEncW[l], convCI[l], convK[l], convK[l], convKtot[l], convKC[l],
        (bf16*)(W + o_cpan[l]), tot);
  }
  for (int l = 0; l < 6; ++l) {
    long tW = (long)(di[l] / 32) * (dd[l] / 16) * 512;
    k_swz_lin<<<(int)((tW + 255) / 256), 256, 0, stream>>>(
        pW[l], di[l], 0, di[l], di[l] / 32, (bf16*)(W + o_wpan[l]), tW);
    long tG = (long)4 * (dd[l] / 16) * 512;
    k_swz_lin<<<(int)((tG + 255) / 256), 256, 0, stream>>>(
        pWg[l], 256, 128, 128, 4, (bf16*)(W + o_gpan[l]), tG);
    k_swz_lin<<<(int)((tG + 255) / 256), 256, 0, stream>>>(
        pWb[l], 256, 128, 128, 4, (bf16*)(W + o_bpan[l]), tG);
  }

  // ---- host schedule math (arctan/cosine logsnr) ----
  const double bbv = atan(exp(-10.0));
  const double aav = atan(exp(10.0)) - bbv;
  auto lsnr = [&](double t) { return -2.0 * log(tan(aav * t + bbv)); };
  TembArgs ta;
  ta.W0 = pTW[0]; ta.b0 = pTB[0]; ta.W1 = pTW[1]; ta.b1 = pTB[1];
  for (int l = 0; l < 6; ++l) { ta.Wg[l] = pWg[l]; ta.bg[l] = pBg[l]; ta.Wb[l] = pWb[l]; ta.dox[l] = dd[l]; ta.doff[l] = doff[l]; }
  ta.gvec = gvec; ta.bvec = bvec;
  for (int i = 0; i < 20; ++i) {
    double lt = lsnr(i / 19.0);
    double cl = lt < -20.0 ? -20.0 : (lt > 20.0 ? 20.0 : lt);
    ta.tt[i] = (float)(atan(exp(-0.5 * cl)) / (0.5 * M_PI) * 1000.0);
  }
  k_temb<<<20, 128, 0, stream>>>(ta);

  // ---- stage 1: encoder (4 chunks of 512 images), fully templated convs ----
  for (int cc = 0; cc < 4; ++cc) {
    const bf16* in0 = xnhwc + (size_t)cc * 512 * 32 * 32 * 3;
    k_conv_gemm<3, 64, 3, 1, 1, 32, 32, 1, 27, 1>
        <<<dim3(8192, 4), 128, 0, stream>>>(in0, (const bf16*)(W + o_cpan[0]), pEncB[0], regB);
    k_conv_gemm<64, 128, 4, 2, 1, 32, 16, 32, 1024, 1>
        <<<dim3(2048, 8), 128, 0, stream>>>(regB, (const bf16*)(W + o_cpan[1]), pEncB[1], regA);
    k_conv_gemm<128, 256, 4, 2, 1, 16, 8, 64, 2048, 1>
        <<<dim3(512, 16), 128, 0, stream>>>(regA, (const bf16*)(W + o_cpan[2]), pEncB[2], regB);
    k_conv_gemm<256, 512, 4, 2, 1, 8, 4, 128, 4096, 1>
        <<<dim3(128, 32), 128, 0, stream>>>(regB, (const bf16*)(W + o_cpan[3]), pEncB[3], regA);
    k_conv_gemm<512, 128, 4, 1, 0, 4, 1, 256, 8192, 0>
        <<<dim3(8, 8), 128, 0, stream>>>(regA, (const bf16*)(W + o_cpan[4]), pEncB[4],
                                         xembbf + (size_t)cc * 512 * 128);
  }

  // ---- stage 2: step-independent xemb parts of gate/bias (once, not 20x) ----
  for (int l = 0; l < 6; ++l) {
    dim3 g(32, dd[l] / 16);
    k_gemm_lin<<<g, 128, 0, stream>>>(xembbf, 128, (const bf16*)(W + o_gpan[l]), 4,
                                      xg + (size_t)2048 * doff[l], nullptr, dd[l],
                                      nullptr, nullptr, nullptr, nullptr, nullptr, 0, 0);
    k_gemm_lin<<<g, 128, 0, stream>>>(xembbf, 128, (const bf16*)(W + o_bpan[l]), 4,
                                      xb + (size_t)2048 * doff[l], nullptr, dd[l],
                                      nullptr, nullptr, nullptr, nullptr, nullptr, 0, 0);
  }

  // ---- stage 3: reverse diffusion ----
  const int NTOT = 2048 * 128;
  k_init_z<<<NTOT / 256, 256, 0, stream>>>(zt, ztb, 0x5EED1000u, NTOT);

  for (int i = 19; i >= 0; --i) {
    const bf16* cur = ztb;
    int curld = 128;
    for (int l = 0; l < 6; ++l) {
      bf16* outb = (l & 1) ? h1 : h0;
      float* outf = (l == 5) ? eps : nullptr;
      dim3 g(32, dd[l] / 16);
      k_gemm_lin<<<g, 128, 0, stream>>>(cur, curld, (const bf16*)(W + o_wpan[l]), di[l] / 32,
                                        outf, outb, dd[l],
                                        xg + (size_t)2048 * doff[l], xb + (size_t)2048 * doff[l],
                                        gvec + (size_t)i * 1408 + doff[l],
                                        bvec + (size_t)i * 1408 + doff[l],
                                        pB[l], 1, (l < 5) ? 1 : 0);
      cur = outb; curld = dd[l];
    }
    double lt = lsnr(i / 19.0);
    double ls = lsnr((i > 0 ? i - 1 : 0) / 19.0);
    float c1 = (float)sqrt(1.0 + exp(-lt));
    float c2 = (float)(1.0 / sqrt(1.0 + exp(lt)));
    double rr = exp(lt - ls);
    double omr = -expm1(lt - ls);
    double ast = sqrt((1.0 + exp(-lt)) / (1.0 + exp(-ls)));
    float ra = (float)(rr * ast);
    float ob = (float)(omr * sqrt(1.0 / (1.0 + exp(-ls))));
    double varr = omr * (1.0 / (1.0 + exp(ls)));
    float stdv = (float)sqrt(varr > 0.0 ? varr : 0.0);
    k_update_z<<<NTOT / 256, 256, 0, stream>>>(eps, zt, ztb, (float*)d_out,
                                               c1, c2, ra, ob, stdv,
                                               (i == 0) ? 1 : 0, 0xC0FFEE00u + (unsigned)i, NTOT);
  }
}
